// Net_53712861003994
// MI455X (gfx1250) — matrix operand order repaired and checked by reading, not compiled, not run on hardware
//
#include <hip/hip_runtime.h>
#include <hip/hip_bf16.h>

typedef __attribute__((ext_vector_type(16))) __bf16 v16bf;
typedef __attribute__((ext_vector_type(8)))  float  v8f;
typedef __attribute__((ext_vector_type(4)))  int    v4i;

#define NN   100000
#define FDIM 512
#define HDIM 128
#define CDIM 64
#define NE   3200000
#define EP   500000
#define ET   1000000

#define AS(n) __attribute__((address_space(n)))

union ABf16 { v16bf v; uint4 q[2]; };

// ---------------- utility kernels ----------------
__global__ void __launch_bounds__(256) k_fill(float* __restrict__ p, float val, int n) {
  int i = blockIdx.x * 256 + threadIdx.x;
  if (i < n) p[i] = val;
}

__global__ void __launch_bounds__(256) k_deg_accum(const int* __restrict__ col,
                                                   float* __restrict__ deg, int e) {
  int i = blockIdx.x * 256 + threadIdx.x;
  if (i < e) atomicAdd(&deg[col[i]], 1.0f);
}

__global__ void __launch_bounds__(256) k_dinv(float* __restrict__ d, int n) {
  int i = blockIdx.x * 256 + threadIdx.x;
  if (i < n) { float v = d[i]; d[i] = v > 0.f ? rsqrtf(v) : 0.f; }
}

// W: [K x Nc] row-major fp32  ->  WT: [Nc x K] bf16 (B-fragment friendly)
__global__ void __launch_bounds__(256) k_cvt_wt(const float* __restrict__ W,
                                                __bf16* __restrict__ WT, int K, int Nc) {
  int i = blockIdx.x * 256 + threadIdx.x;
  if (i < K * Nc) {
    int k = i / Nc, n = i % Nc;
    WT[(size_t)n * K + k] = (__bf16)W[i];
  }
}

// ---------------- WMMA GEMM: Y[100000 x 128] = X[100000 x 512] * W1 ----------------
__global__ void __launch_bounds__(128) k_gemm1(const float* __restrict__ X,
                                               const __bf16* __restrict__ WT,
                                               float* __restrict__ Y) {
  __shared__ __attribute__((aligned(32))) __bf16 As[16 * FDIM];
  const int tid = threadIdx.x;
  const size_t m0 = (size_t)blockIdx.x * 16;

  // stage 16x512 fp32 tile -> bf16 LDS (conversion forces the VALU path here)
  const float4* xs = (const float4*)(X + m0 * FDIM);
  for (int i = tid; i < 16 * FDIM / 4; i += 128) {
    float4 v = xs[i];
    int b = i * 4;
    As[b + 0] = (__bf16)v.x; As[b + 1] = (__bf16)v.y;
    As[b + 2] = (__bf16)v.z; As[b + 3] = (__bf16)v.w;
  }
  __syncthreads();

  const int lane = tid & 31, wave = tid >> 5;
  const int n0    = wave * 32;
  const int rowA  = lane & 15;
  const int koffA = (lane >> 4) * 16;  // A frag, contiguous k order used here: lanes 0-15 K 0-15, lanes 16-31 K 16-31
  const int koffB = (lane >> 4) * 16;  // B frag, same contiguous k order as A: lanes 0-15 K 0-15, lanes 16-31 K 16-31
  const int cB0   = n0 + (lane & 15);
  const int cB1   = cB0 + 16;

  v8f acc0 = {0,0,0,0,0,0,0,0}, acc1 = {0,0,0,0,0,0,0,0};
  for (int k = 0; k < FDIM; k += 32) {
    ABf16 a;
    a.q[0] = *(const uint4*)&As[rowA * FDIM + k + koffA];
    a.q[1] = *(const uint4*)&As[rowA * FDIM + k + koffA + 8];
    v16bf b0 = *(const v16bf*)&WT[(size_t)cB0 * FDIM + k + koffB];
    v16bf b1 = *(const v16bf*)&WT[(size_t)cB1 * FDIM + k + koffB];
    acc0 = __builtin_amdgcn_wmma_f32_16x16x32_bf16(false, a.v, false, b0, (short)0, acc0, false, false);
    acc1 = __builtin_amdgcn_wmma_f32_16x16x32_bf16(false, a.v, false, b1, (short)0, acc1, false, false);
  }

  const int mo = (int)m0 + (lane >> 4) * 8;  // C/D layout: lanes>=16 hold M+8
  const int c0 = n0 + (lane & 15);
#pragma unroll
  for (int v = 0; v < 8; ++v) {
    Y[(size_t)(mo + v) * HDIM + c0]      = acc0[v];
    Y[(size_t)(mo + v) * HDIM + c0 + 16] = acc1[v];
  }
}

// ---------------- WMMA GEMM: Y[100000 x 64] = H1[100000 x 128](bf16) * W2 ----------------
__global__ void __launch_bounds__(128) k_gemm2(const __bf16* __restrict__ H1,
                                               const __bf16* __restrict__ WT,
                                               float* __restrict__ Y) {
  __shared__ __attribute__((aligned(32))) __bf16 As[16 * HDIM];
  const int tid = threadIdx.x;
  const size_t m0 = (size_t)blockIdx.x * 16;

  // stage 16x128 bf16 tile: contiguous 4KB -> use CDNA5 async global->LDS when available
#if __has_builtin(__builtin_amdgcn_global_load_async_to_lds_b128)
  {
    AS(1) v4i* gsrc = (AS(1) v4i*)(uintptr_t)(H1 + m0 * HDIM);
    AS(3) v4i* ldst = (AS(3) v4i*)(unsigned)(uintptr_t)(&As[0]);
    for (int i = tid; i < (16 * HDIM * 2) / 16; i += 128)
      __builtin_amdgcn_global_load_async_to_lds_b128(gsrc + i, ldst + i, 0, 0);
#if __has_builtin(__builtin_amdgcn_s_wait_asynccnt)
    __builtin_amdgcn_s_wait_asynccnt(0);
#else
    asm volatile("s_wait_asynccnt 0x0" ::: "memory");
#endif
  }
#else
  {
    const uint4* src = (const uint4*)(H1 + m0 * HDIM);
    uint4* dst = (uint4*)As;
    for (int i = tid; i < (16 * HDIM * 2) / 16; i += 128) dst[i] = src[i];
  }
#endif
  __syncthreads();

  const int lane = tid & 31, wave = tid >> 5;
  const int n0    = wave * 16;
  const int rowA  = lane & 15;
  const int koffA = (lane >> 4) * 16;
  const int koffB = (lane >> 4) * 16;
  const int cB    = n0 + (lane & 15);

  v8f acc = {0,0,0,0,0,0,0,0};
  for (int k = 0; k < HDIM; k += 32) {
    ABf16 a;
    a.q[0] = *(const uint4*)&As[rowA * HDIM + k + koffA];
    a.q[1] = *(const uint4*)&As[rowA * HDIM + k + koffA + 8];
    v16bf b = *(const v16bf*)&WT[(size_t)cB * HDIM + k + koffB];
    acc = __builtin_amdgcn_wmma_f32_16x16x32_bf16(false, a.v, false, b, (short)0, acc, false, false);
  }

  const int mo = (int)m0 + (lane >> 4) * 8;
  const int c0 = n0 + (lane & 15);
#pragma unroll
  for (int v = 0; v < 8; ++v) Y[(size_t)(mo + v) * CDIM + c0] = acc[v];
}

// ---------------- scatter-add message passing ----------------
// One wave32 per 4 edges; prefetch the next edge's source row (global_prefetch_b8)
// while the current 512B/256B gather + atomics are in flight.
__global__ void __launch_bounds__(256) k_scatter128(const float* __restrict__ h,
                                                    const int* __restrict__ rows,
                                                    const int* __restrict__ cols,
                                                    const float* __restrict__ dinv,
                                                    float* __restrict__ out, int e) {
  int wid  = (blockIdx.x * 256 + threadIdx.x) >> 5;
  int lane = threadIdx.x & 31;
  int base = wid * 4;
  if (base >= e) return;
  int r = rows[base], c = cols[base];
#pragma unroll
  for (int j = 0; j < 4; ++j) {
    int rn = 0, cn = 0;
    if (j < 3) {
      rn = rows[base + j + 1];
      cn = cols[base + j + 1];
      __builtin_prefetch(h + (size_t)rn * HDIM + lane * 4, 0, 3);
    }
    float nrm = dinv[r] * dinv[c];
    float4 v = ((const float4*)(h + (size_t)r * HDIM))[lane];
    float* o = out + (size_t)c * HDIM + lane * 4;
    atomicAdd(o + 0, v.x * nrm); atomicAdd(o + 1, v.y * nrm);
    atomicAdd(o + 2, v.z * nrm); atomicAdd(o + 3, v.w * nrm);
    r = rn; c = cn;
  }
}

__global__ void __launch_bounds__(256) k_scatter64(const float* __restrict__ h,
                                                   const int* __restrict__ rows,
                                                   const int* __restrict__ cols,
                                                   const float* __restrict__ dinv,
                                                   float* __restrict__ out, int e) {
  int wid  = (blockIdx.x * 256 + threadIdx.x) >> 5;
  int lane = threadIdx.x & 31;
  int base = wid * 4;
  if (base >= e) return;
  int r = rows[base], c = cols[base];
#pragma unroll
  for (int j = 0; j < 4; ++j) {
    int rn = 0, cn = 0;
    if (j < 3) {
      rn = rows[base + j + 1];
      cn = cols[base + j + 1];
      __builtin_prefetch(h + (size_t)rn * CDIM + lane * 2, 0, 3);
    }
    float nrm = dinv[r] * dinv[c];
    float2 v = ((const float2*)(h + (size_t)r * CDIM))[lane];
    float* o = out + (size_t)c * CDIM + lane * 2;
    atomicAdd(o + 0, v.x * nrm); atomicAdd(o + 1, v.y * nrm);
    r = rn; c = cn;
  }
}

// self-loop + bias + relu -> bf16 for next GEMM
__global__ void __launch_bounds__(256) k_final1(const float* __restrict__ agg,
                                                const float* __restrict__ hW,
                                                const float* __restrict__ dinv,
                                                const float* __restrict__ bias,
                                                __bf16* __restrict__ h1) {
  int i = blockIdx.x * 256 + threadIdx.x;
  if (i >= NN * HDIM) return;
  int node = i >> 7, c = i & 127;
  float di = dinv[node];
  float v = agg[i] + hW[i] * di * di + bias[c];
  h1[i] = (__bf16)fmaxf(v, 0.f);
}

// self-loop + bias (no relu), in-place into agg -> h2
__global__ void __launch_bounds__(256) k_final2(float* __restrict__ agg,
                                                const float* __restrict__ hW,
                                                const float* __restrict__ dinv,
                                                const float* __restrict__ bias) {
  int i = blockIdx.x * 256 + threadIdx.x;
  if (i >= NN * CDIM) return;
  int node = i >> 6, c = i & 63;
  float di = dinv[node];
  agg[i] = agg[i] + hW[i] * di * di + bias[c];
}

// ---------------- edge scoring (one wave32 per edge) ----------------
__global__ void __launch_bounds__(256) k_score(const float* __restrict__ h2,
                                               const int* __restrict__ pos,
                                               const int* __restrict__ neg,
                                               const float* __restrict__ w,
                                               const float* __restrict__ sb,
                                               float* __restrict__ out,
                                               float* __restrict__ lacc) {
  __shared__ float red[8];
  int gw = (blockIdx.x * 256 + threadIdx.x) >> 5;
  int lane = threadIdx.x & 31;
  float s = 0.f;
  if (gw < ET) {
    int src, dst;
    if (gw < EP) { src = pos[gw]; dst = pos[EP + gw]; }
    else         { int e2 = gw - EP; src = neg[e2]; dst = neg[EP + e2]; }
    float2 a  = ((const float2*)(h2 + (size_t)src * CDIM))[lane];
    float2 b  = ((const float2*)(h2 + (size_t)dst * CDIM))[lane];
    float2 ww = ((const float2*)w)[lane];
    s = (a.x - b.x) * ww.x + (a.y - b.y) * ww.y;
#pragma unroll
    for (int off = 16; off > 0; off >>= 1) s += __shfl_xor(s, off, 32);
    if (lane == 0) out[gw] = fmaxf(s + sb[0], 0.f);
  }
  if (lane == 0) red[threadIdx.x >> 5] = s;
  __syncthreads();
  if (threadIdx.x == 0) {
    float t = red[0] + red[1] + red[2] + red[3] + red[4] + red[5] + red[6] + red[7];
    atomicAdd(lacc, t);
  }
}

__global__ void k_loss(const float* __restrict__ lacc, float* __restrict__ outloss) {
  if (threadIdx.x == 0 && blockIdx.x == 0) outloss[0] = lacc[0] * (1.0f / (float)ET);
}

// ---------------- driver ----------------
extern "C" void kernel_launch(void* const* d_in, const int* in_sizes, int n_in,
                              void* d_out, int out_size, void* d_ws, size_t ws_size,
                              hipStream_t stream) {
  (void)in_sizes; (void)n_in; (void)out_size; (void)ws_size;
  const float* x   = (const float*)d_in[0];
  const int*   ei  = (const int*)d_in[1];
  const int*   pei = (const int*)d_in[2];
  const int*   nei = (const int*)d_in[3];
  const float* W1  = (const float*)d_in[4];
  const float* b1  = (const float*)d_in[5];
  const float* W2  = (const float*)d_in[6];
  const float* b2  = (const float*)d_in[7];
  const float* sw  = (const float*)d_in[8];
  const float* sb  = (const float*)d_in[9];
  float* out = (float*)d_out;

  char* ws = (char*)d_ws;
  size_t off = 0;
  auto carve = [&](size_t bytes) -> char* {
    char* p = ws + off;
    off += (bytes + 255) & ~(size_t)255;
    return p;
  };
  float*  dinv = (float*)carve((size_t)NN * 4);
  __bf16* W1T  = (__bf16*)carve((size_t)HDIM * FDIM * 2);
  __bf16* W2T  = (__bf16*)carve((size_t)CDIM * HDIM * 2);
  float*  hW1  = (float*)carve((size_t)NN * HDIM * 4);
  float*  agg1 = (float*)carve((size_t)NN * HDIM * 4);
  __bf16* h1   = (__bf16*)carve((size_t)NN * HDIM * 2);
  float*  hW2  = (float*)carve((size_t)NN * CDIM * 4);
  float*  agg2 = (float*)carve((size_t)NN * CDIM * 4);
  float*  lacc = (float*)carve(256);

  const int* rows = ei;       // edge_index[0] = sources
  const int* cols = ei + NE;  // edge_index[1] = targets

  // degrees (self-loop = init 1.0) -> dinv
  k_fill<<<(NN + 255) / 256, 256, 0, stream>>>(dinv, 1.0f, NN);
  k_deg_accum<<<(NE + 255) / 256, 256, 0, stream>>>(cols, dinv, NE);
  k_dinv<<<(NN + 255) / 256, 256, 0, stream>>>(dinv, NN);

  // weight transpose + bf16 convert
  k_cvt_wt<<<(FDIM * HDIM + 255) / 256, 256, 0, stream>>>(W1, W1T, FDIM, HDIM);
  k_cvt_wt<<<(HDIM * CDIM + 255) / 256, 256, 0, stream>>>(W2, W2T, HDIM, CDIM);

  // zero accumulators (every launch: graph-replay safe)
  k_fill<<<(NN * HDIM + 255) / 256, 256, 0, stream>>>(agg1, 0.0f, NN * HDIM);
  k_fill<<<(NN * CDIM + 255) / 256, 256, 0, stream>>>(agg2, 0.0f, NN * CDIM);
  k_fill<<<1, 256, 0, stream>>>(lacc, 0.0f, 1);

  // layer 1
  k_gemm1<<<NN / 16, 128, 0, stream>>>(x, W1T, hW1);
  k_scatter128<<<NE / 32, 256, 0, stream>>>(hW1, rows, cols, dinv, agg1, NE);
  k_final1<<<(NN * HDIM + 255) / 256, 256, 0, stream>>>(agg1, hW1, dinv, b1, h1);

  // layer 2
  k_gemm2<<<NN / 16, 128, 0, stream>>>(h1, W2T, hW2);
  k_scatter64<<<NE / 32, 256, 0, stream>>>(hW2, rows, cols, dinv, agg2, NE);
  k_final2<<<(NN * CDIM + 255) / 256, 256, 0, stream>>>(agg2, hW2, dinv, b2);

  // edge scoring + loss
  k_score<<<ET / 8, 256, 0, stream>>>(agg2, pei, nei, sw, sb, out, lacc);
  k_loss<<<1, 1, 0, stream>>>(lacc, out + ET);
}
